// GCN_alignment_iia_47502338294134
// MI455X (gfx1250) — matrix operand order repaired and checked by reading, not compiled, not run on hardware
//
#include <hip/hip_runtime.h>
#include <cmath>

// ---------------------------------------------------------------------------
// Types for CDNA5 WMMA (wave32): 16x16x32 bf16 -> f32 accumulator
// ---------------------------------------------------------------------------
typedef __attribute__((ext_vector_type(16))) __bf16 v16bf;
typedef __attribute__((ext_vector_type(8)))  float  v8f;

union FragBF16 { uint4 q[2]; v16bf v; };   // 32 bytes: one WMMA A/B operand

#define LDS_STRIDE 48   // halves per LDS tile row: 32 data + 16 pad (96B rows, 16B-aligned)

// Round-half-up bf16 pack of two f32 (differs from RNE only on exact ties).
__device__ __forceinline__ unsigned pack2_bf16(float x, float y) {
  unsigned a = __float_as_uint(x) + 0x8000u;
  unsigned b = __float_as_uint(y) + 0x8000u;
  return (a >> 16) | (b & 0xFFFF0000u);
}

__device__ __forceinline__ unsigned short f32_to_bf16(float f) {
  unsigned u = __float_as_uint(f);
  return (unsigned short)((u + 0x7FFFu + ((u >> 16) & 1u)) >> 16);  // RNE
}

// ---------------------------------------------------------------------------
// embedding [8192][256] f32  ->  embT [256][8192] bf16   (LDS tiled transpose)
// ---------------------------------------------------------------------------
__global__ __launch_bounds__(256)
void k_transpose_bf16(const float* __restrict__ E, unsigned short* __restrict__ ET) {
  __shared__ float tile[32][33];
  const int tx = threadIdx.x, ty = threadIdx.y;
  const int kb = blockIdx.x * 32;      // K base (rows of E)
  const int hb = blockIdx.y * 32;      // H base (cols of E)
#pragma unroll
  for (int i = 0; i < 32; i += 8)
    tile[ty + i][tx] = E[(size_t)(kb + ty + i) * 256 + hb + tx];
  __syncthreads();
#pragma unroll
  for (int i = 0; i < 32; i += 8)
    ET[(size_t)(hb + ty + i) * 8192 + kb + tx] = f32_to_bf16(tile[tx][ty + i]);
}

// ---------------------------------------------------------------------------
// Build WkT [384][512] bf16 from conv kernels (H-padding collapses the conv)
// ---------------------------------------------------------------------------
__global__ __launch_bounds__(256)
void k_build_wkt(const float* __restrict__ k1, const float* __restrict__ k2,
                 unsigned short* __restrict__ WkT) {
  int t = blockIdx.x * 256 + threadIdx.x;
  if (t >= 384 * 512) return;
  int c = t >> 9, j = t & 511;
  float v;
  if (c < 128)      v = k1[c * 514 + j + 1];
  else if (c < 256) v = k2[(c - 128) * 1028 + j + 1];
  else              v = k2[(c - 256) * 1028 + 514 + j + 1];
  WkT[t] = f32_to_bf16(v);
}

// ---------------------------------------------------------------------------
// T [2048][512] f32 from K-split partial sums: layer = LP[0] + LP[1]
// T = [ (rQ-rA)^2 , rQ*rA ];  rQ = layer rows 0..2047, rA = rows 2048..4095
// ---------------------------------------------------------------------------
__global__ __launch_bounds__(256)
void k_build_T(const float* __restrict__ LP, float* __restrict__ T) {
  int t = blockIdx.x * 256 + threadIdx.x;
  if (t >= 2048 * 256) return;
  int b = t >> 8, h = t & 255;
  const size_t part = (size_t)4096 * 256;
  float q = LP[(size_t)b * 256 + h]          + LP[part + (size_t)b * 256 + h];
  float a = LP[(size_t)(b + 2048) * 256 + h] + LP[part + (size_t)(b + 2048) * 256 + h];
  float d = q - a;
  T[(size_t)b * 512 + h]       = d * d;
  T[(size_t)b * 512 + 256 + h] = q * a;
}

// ---------------------------------------------------------------------------
// bf16 WMMA GEMM, double-buffered LDS, async B staging, row gather, K-split.
//   C[kc][M][ldc] (f32) = gatherRows(Asrc)[M][kbeg:kend] (f32->bf16) @ BT^T
// Block: 256 threads = 8 waves as 2(M) x 4(N); tile 32(M) x 64*NWSUB(N); K-step 32.
// Pipeline (one barrier per K-step):
//   stage block k+32 -> buffers[nxt] (B: global_load_async_to_lds_b128,
//   A: f32 regs pipelined one step ahead, packed to bf16, ds_store)
//   compute block k from buffers[cur] (ds_load_b128 frags + NWSUB wmma)
//   s_wait_asynccnt 0 ; barrier ; swap
// Race-freedom: writes to buffer nxt at step k happen after barrier(k-1);
// the last reads of that buffer (step k-1, as cur) completed before each wave
// signaled barrier(k-1) (dscnt/asynccnt drained before the signal).
// ---------------------------------------------------------------------------
template <int NWSUB>
__global__ __launch_bounds__(256)
void k_gemm_bf16(const float* __restrict__ Asrc, int lda,
                 const int* __restrict__ idx0, const int* __restrict__ idx1, int split,
                 const unsigned short* __restrict__ BT, int ldb,
                 float* __restrict__ Cbase, int ldc,
                 int M, int K, int nksplit) {
  constexpr int NB = 64 * NWSUB;                 // block N tile
  __shared__ unsigned short As[2][32 * LDS_STRIDE];
  __shared__ unsigned short Bs[2][NB * LDS_STRIDE];

  const int tid   = threadIdx.x;
  const int wave  = tid >> 5;
  const int lane  = tid & 31;
  const int waveM = wave >> 2;                   // 0..1
  const int waveN = wave & 3;                    // 0..3
  const int hi = lane >> 4;                      // half-wave: K-slab select in fragments
  const int lx = lane & 15;

  const int m0 = blockIdx.x * 32;
  const int n0 = blockIdx.y * NB;
  const int kc = blockIdx.z;
  const int kchunk = K / nksplit;
  const int kbeg = kc * kchunk, kend = kbeg + kchunk;
  float* C = Cbase + (size_t)kc * M * ldc;

  // Fixed A staging slot: one float4 per thread per K-step (gather resolved once).
  const int ar  = tid >> 3;                      // 0..31: tile row
  const int ac4 = tid & 7;                       // 0..7:  float4 column
  int agrow = m0 + ar;
  if (idx0) agrow = (agrow < split) ? idx0[agrow] : idx1[agrow - split];
  const float* aptr = Asrc + (size_t)agrow * lda + ac4 * 4;

  // Fixed B staging slots: NWSUB 16B async copies per thread per K-step.
  const int br  = tid >> 2;                      // 0..63: row base (stride 64 rows)
  const int bc8 = tid & 3;                       // 16B chunk within the 32-K row
  const unsigned short* bptr = BT + (size_t)(n0 + br) * ldb + bc8 * 8;

  unsigned short* asl[2];
  unsigned bdst[2];
#pragma unroll
  for (int p = 0; p < 2; ++p) {
    asl[p]  = &As[p][ar * LDS_STRIDE + ac4 * 4];
    bdst[p] = (unsigned)(size_t)(void*)&Bs[p][br * LDS_STRIDE + bc8 * 8];  // LDS byte offset
  }

  auto stageB = [&](int p, int k) {
#pragma unroll
    for (int i = 0; i < NWSUB; ++i) {
      const unsigned short* gp = bptr + (size_t)i * 64 * ldb + k;
      unsigned l = bdst[p] + i * 64 * LDS_STRIDE * 2;
      asm volatile("global_load_async_to_lds_b128 %0, %1, off"
                   :: "v"(l), "v"(gp) : "memory");
    }
  };
  auto stageA = [&](int p, const float4& f) {
    *(uint2*)asl[p] = make_uint2(pack2_bf16(f.x, f.y), pack2_bf16(f.z, f.w));
  };

  v8f acc[NWSUB];
#pragma unroll
  for (int j = 0; j < NWSUB; ++j) acc[j] = v8f{};

  // ---- prologue: fill buffer 0 with block kbeg; pipeline A regs one step ----
  float4 f0 = *(const float4*)(aptr + kbeg);
  stageB(0, kbeg);
  float4 fnext = (kbeg + 32 < kend) ? *(const float4*)(aptr + kbeg + 32) : f0;
  stageA(0, f0);
  asm volatile("s_wait_asynccnt 0x0" ::: "memory");
  __syncthreads();

  int cur = 0;
  for (int k0 = kbeg; k0 < kend; k0 += 32) {
    const int nxt = cur ^ 1;
    const bool hasNext = (k0 + 32 < kend);
    if (hasNext) {
      stageB(nxt, k0 + 32);                      // async DMA into the idle buffer
      stageA(nxt, fnext);
      if (k0 + 64 < kend) {
        fnext = *(const float4*)(aptr + k0 + 64);
        __builtin_prefetch((const void*)(aptr + k0 + 128), 0, 0);  // global_prefetch_b8
      }
    }
    // ---- compute block k0 from buffers[cur] ----
    // A fragment (ISA 16-bit A 16x32 layout): e 0..7 -> K=hi*8+e ; e 8..15 -> K=16+hi*8+e
    FragBF16 a;
    a.q[0] = *(const uint4*)(&As[cur][(waveM * 16 + lx) * LDS_STRIDE + hi * 8]);
    a.q[1] = *(const uint4*)(&As[cur][(waveM * 16 + lx) * LDS_STRIDE + 16 + hi * 8]);
#pragma unroll
    for (int j = 0; j < NWSUB; ++j) {
      // B fragment: N = lane%16, e 0..7 -> K=hi*8+e ; e 8..15 -> K=16+hi*8+e (Bs is N-major)
      const int bn = (waveN * NWSUB + j) * 16 + lx;
      FragBF16 b;
      b.q[0] = *(const uint4*)(&Bs[cur][bn * LDS_STRIDE + hi * 8]);
      b.q[1] = *(const uint4*)(&Bs[cur][bn * LDS_STRIDE + 16 + hi * 8]);
      acc[j] = __builtin_amdgcn_wmma_f32_16x16x32_bf16(
          false, a.v, false, b.v, (short)0, acc[j], false, false);
    }
    asm volatile("s_wait_asynccnt 0x0" ::: "memory");
    __syncthreads();
    cur = nxt;
  }

  // ---- epilogue (D layout: m = 8*hi + vgpr, n = lane%16); all dims divide ----
#pragma unroll
  for (int j = 0; j < NWSUB; ++j) {
    const int col = n0 + (waveN * NWSUB + j) * 16 + lx;
#pragma unroll
    for (int r = 0; r < 8; ++r) {
      const int row = m0 + waveM * 16 + hi * 8 + r;
      C[(size_t)row * ldc + col] = acc[j][r];
    }
  }
}

// ---------------------------------------------------------------------------
// Head: one wave per batch row; maxpool/relu + folded dense + log-softmax.
// ---------------------------------------------------------------------------
__global__ __launch_bounds__(256)
void k_finalize(const float* __restrict__ S, const float* __restrict__ b1,
                const float* __restrict__ b2, const float* __restrict__ dw,
                const float* __restrict__ db, float* __restrict__ out) {
  const int lane = threadIdx.x & 31;
  const int b = blockIdx.x * 8 + (threadIdx.x >> 5);
  const float* Sb = S + (size_t)b * 384;
  float p0 = 0.f, p1 = 0.f;
#pragma unroll
  for (int i = 0; i < 4; ++i) {
    int c = lane + i * 32;                       // 0..127
    float s  = Sb[c];
    float bb = b1[c];
    float e1 = fmaxf(fmaxf(bb, s + bb), 0.f);
    p0 += e1 * (dw[c]       + dw[c + 256]);
    p1 += e1 * (dw[512 + c] + dw[768 + c]);
    float e2 = fmaxf(fmaxf(Sb[128 + c], Sb[256 + c]) + b2[c], 0.f);
    p0 += e2 * (dw[128 + c]       + dw[384 + c]);
    p1 += e2 * (dw[512 + 128 + c] + dw[512 + 384 + c]);
  }
#pragma unroll
  for (int off = 16; off; off >>= 1) {
    p0 += __shfl_xor(p0, off, 32);
    p1 += __shfl_xor(p1, off, 32);
  }
  if (lane == 0) {
    float s0 = p0 + db[0], s1 = p1 + db[1];
    float mx = fmaxf(s0, s1);
    float l  = mx + logf(expf(s0 - mx) + expf(s1 - mx));
    out[(size_t)b * 2 + 0] = s0 - l;
    out[(size_t)b * 2 + 1] = s1 - l;
  }
}

// ---------------------------------------------------------------------------
// Workspace (12.4 MB, safe intra-call aliasing; every byte read is written
// earlier in the same call):
//   [0, 4MB):     embT bf16  -> reused as T [2048][512] f32 after GEMM1
//   [4MB, 12MB):  layer partials [2][4096][256] f32 -> reused as S after build_T
//   [12MB, ...):  WkT [384][512] bf16
// ---------------------------------------------------------------------------
extern "C" void kernel_launch(void* const* d_in, const int* in_sizes, int n_in,
                              void* d_out, int out_size, void* d_ws, size_t ws_size,
                              hipStream_t stream) {
  const float* emb = (const float*)d_in[0];   // [8192][256]
  const float* A   = (const float*)d_in[1];   // [8192][8192]
  const float* k1  = (const float*)d_in[2];   // [128][1][1][514]
  const float* b1  = (const float*)d_in[3];   // [128]
  const float* k2  = (const float*)d_in[4];   // [128][1][2][514]
  const float* b2  = (const float*)d_in[5];   // [128]
  const float* dw  = (const float*)d_in[6];   // [2][512]
  const float* db  = (const float*)d_in[7];   // [2]
  const int*   bx  = (const int*)d_in[8];     // [2048]
  const int*   by  = (const int*)d_in[9];     // [2048]
  float* out = (float*)d_out;                 // [2048][2]

  char* ws = (char*)d_ws;
  unsigned short* embT   = (unsigned short*)(ws);               // 4 MB
  float*          T      = (float*)(ws);                        // aliases embT (after GEMM1)
  float*          layerP = (float*)(ws + (4u << 20));           // 8 MB: 2 K-split partials
  float*          S      = (float*)(ws + (4u << 20));           // aliases layerP (after build_T)
  unsigned short* WkT    = (unsigned short*)(ws + (12u << 20)); // 384 KB

  // 1) embedding -> embT (bf16, K-contiguous for GEMM B-tiles)
  k_transpose_bf16<<<dim3(8192 / 32, 256 / 32), dim3(32, 8), 0, stream>>>(emb, embT);
  // 2) conv weights -> WkT (bf16)
  k_build_wkt<<<(384 * 512 + 255) / 256, 256, 0, stream>>>(k1, k2, WkT);
  // 3) gathered GCN GEMM, full-N strip + 2-way K split (A streamed from HBM once)
  k_gemm_bf16<4><<<dim3(4096 / 32, 1, 2), 256, 0, stream>>>(
      A, 8192, bx, by, 2048, embT, 8192, layerP, 256, 4096, 8192, 2);
  // 4) T = [ (rQ-rA)^2 , rQ*rA ] from the two partials
  k_build_T<<<(2048 * 256 + 255) / 256, 256, 0, stream>>>(layerP, T);
  // 5) head GEMM: S = T @ Wk   (M=2048, N=384, K=512)
  k_gemm_bf16<1><<<dim3(2048 / 32, 6, 1), 256, 0, stream>>>(
      T, 512, nullptr, nullptr, 0, WkT, 512, S, 384, 2048, 512, 1);
  // 6) maxpool/relu/dense/log-softmax
  k_finalize<<<2048 / 8, 256, 0, stream>>>(S, b1, b2, dw, db, out);
}
